// DTWModel_36438502539737
// MI455X (gfx1250) — matrix operand order repaired and checked by reading, not compiled, not run on hardware
//
#include <hip/hip_runtime.h>
#include <hip/hip_bf16.h>

typedef _Float16 v16h __attribute__((ext_vector_type(16)));
typedef float    v8f  __attribute__((ext_vector_type(8)));

constexpr int   Bn    = 8;
constexpr int   Tn    = 512;
constexpr int   Hn    = 128;
constexpr int   CQn   = 126;
constexpr int   CKn   = 80;
constexpr float SLOPE = 0.2f;
constexpr float INFF  = 1e30f;

__device__ __forceinline__ int reflect1(int s) {
  if (s < 0) s = -s;                       // -1 -> 1
  if (s > Tn - 1) s = 2 * (Tn - 1) - s;    // 512 -> 510
  return s;
}

// ---------------- ConvBlock: reflect pad, K=3, bias, leaky relu -------------
__global__ void conv_block_k(const float* __restrict__ x, const float* __restrict__ w,
                             const float* __restrict__ bias, float* __restrict__ y,
                             int Cin, int Cout) {
  int idx = blockIdx.x * blockDim.x + threadIdx.x;
  int total = Bn * Cout * Tn;
  if (idx >= total) return;
  int t  = idx % Tn;
  int co = (idx / Tn) % Cout;
  int b  = idx / (Tn * Cout);
  const float* xb = x + (size_t)b * Cin * Tn;
  const float* wc = w + (size_t)co * Cin * 3;
  int tm = reflect1(t - 1);
  int tp = reflect1(t + 1);
  float acc = bias[co];
  for (int ci = 0; ci < Cin; ++ci) {
    const float* xc = xb + (size_t)ci * Tn;
    acc = fmaf(wc[ci * 3 + 0], xc[tm], acc);
    acc = fmaf(wc[ci * 3 + 1], xc[t],  acc);
    acc = fmaf(wc[ci * 3 + 2], xc[tp], acc);
  }
  y[idx] = (acc > 0.0f) ? acc : SLOPE * acc;
}

// ---------------- squared row norms of the f32 latents ----------------------
__global__ void norms_k(const float* __restrict__ q, const float* __restrict__ k,
                        float* __restrict__ qnorm, float* __restrict__ knorm) {
  int idx = blockIdx.x * blockDim.x + threadIdx.x;   // over Bn*Tn
  if (idx >= Bn * Tn) return;
  int t = idx % Tn, b = idx / Tn;
  float sq = 0.f, sk = 0.f;
  for (int h = 0; h < Hn; ++h) {
    size_t o = ((size_t)b * Hn + h) * Tn + t;
    float qv = q[o], kv = k[o];
    sq = fmaf(qv, qv, sq);
    sk = fmaf(kv, kv, sk);
  }
  qnorm[idx] = sq;
  knorm[idx] = sk;
}

// ------- swizzle f32 latent [B,H,T] -> fragment-ready f16 blocks ------------
// Block layout: frag[(b*32 + tile)*4 + kc] = 32 lanes x 16 halves (contiguous,
// k = 16*(lane/16) + element in both A and B images, a valid WMMA operand order).
// mode 0 = A-matrix image (k latent), mode 1 = B-matrix image (q latent).
__global__ void swizzle_frag_k(const float* __restrict__ lat, _Float16* __restrict__ out,
                               int mode) {
  int idx = blockIdx.x * blockDim.x + threadIdx.x;   // Bn*32*4*32 = 32768
  if (idx >= Bn * 32 * 4 * 32) return;
  int lane = idx & 31;
  int kc   = (idx >> 5) & 3;
  int tile = (idx >> 7) & 31;
  int b    = idx >> 12;
  int col  = tile * 16 + (lane & 15);
  int hiK  = (lane & 16) ? 1 : 0;
  const float* lb = lat + (size_t)b * Hn * Tn;
  union Frag { v16h v; _Float16 e[16]; } f;
#pragma unroll
  for (int hi = 0; hi < 16; ++hi) {
    int h;
    if (mode == 0)  // A image: lane<16 -> K {0..15}; lane>=16 -> K {16..31}
      h = kc * 32 + (hiK ? 16 : 0) + hi;
    else            // B image: lane<16 -> K {0..15}; lane>=16 -> K {16..31}
      h = kc * 32 + (hiK ? 16 : 0) + hi;
    f.e[hi] = (_Float16)lb[(size_t)h * Tn + col];
  }
  *(v16h*)(out + (size_t)idx * 16) = f.v;
}

// ------------- distance matrix via v_wmma_f32_16x16x32_f16 ------------------
// One wave computes a 16(M) x 64(N) strip: A fragment reused across 4 B tiles.
// dist[b][m][n] = sqrt(|k_m|^2 + |q_n|^2 - 2 * <k_m, q_n>)
__global__ void __launch_bounds__(128)
dist_wmma_k(const _Float16* __restrict__ kswz, const _Float16* __restrict__ qswz,
            const float* __restrict__ knorm, const float* __restrict__ qnorm,
            float* __restrict__ dist) {
  int wave = blockIdx.x * (blockDim.x >> 5) + (threadIdx.x >> 5); // 0..2047
  int lane = threadIdx.x & 31;
  int ns = wave & 7;          // N strip of 4 tiles
  int tm = (wave >> 3) & 31;  // M tile
  int b  = wave >> 8;
  const _Float16* ka = kswz + (((size_t)b * 32 + tm) * 4) * 512;
  const _Float16* qa = qswz + (((size_t)b * 32 + ns * 4) * 4) * 512;

  v8f acc0 = {}, acc1 = {}, acc2 = {}, acc3 = {};
  union Frag { v16h v; _Float16 e[16]; };
#pragma unroll
  for (int kc = 0; kc < 4; ++kc) {
    Frag Af;
    Af.v = *(const v16h*)(ka + ((size_t)kc * 32 + lane) * 16);
    Frag B0, B1, B2, B3;
    B0.v = *(const v16h*)(qa + (((size_t)0 * 4 + kc) * 32 + lane) * 16);
    B1.v = *(const v16h*)(qa + (((size_t)1 * 4 + kc) * 32 + lane) * 16);
    B2.v = *(const v16h*)(qa + (((size_t)2 * 4 + kc) * 32 + lane) * 16);
    B3.v = *(const v16h*)(qa + (((size_t)3 * 4 + kc) * 32 + lane) * 16);
    acc0 = __builtin_amdgcn_wmma_f32_16x16x32_f16(false, Af.v, false, B0.v, (short)0, acc0, false, false);
    acc1 = __builtin_amdgcn_wmma_f32_16x16x32_f16(false, Af.v, false, B1.v, (short)0, acc1, false, false);
    acc2 = __builtin_amdgcn_wmma_f32_16x16x32_f16(false, Af.v, false, B2.v, (short)0, acc2, false, false);
    acc3 = __builtin_amdgcn_wmma_f32_16x16x32_f16(false, Af.v, false, B3.v, (short)0, acc3, false, false);
  }

  int mBase = tm * 16 + ((lane & 16) ? 8 : 0);
  int lm = lane & 15;
#pragma unroll
  for (int s = 0; s < 4; ++s) {
    v8f acc = (s == 0) ? acc0 : (s == 1) ? acc1 : (s == 2) ? acc2 : acc3;
    int n = (ns * 4 + s) * 16 + lm;
    float qn = qnorm[b * Tn + n];
#pragma unroll
    for (int r = 0; r < 8; ++r) {
      int m = mBase + r;
      float d2 = knorm[b * Tn + m] + qn - 2.0f * acc[r];
      dist[((size_t)b * Tn + m) * Tn + n] = sqrtf(fmaxf(d2, 0.0f));
    }
  }
}

// ---------------- global min/max (two-pass, deterministic) ------------------
__global__ void minmax_partial_k(const float* __restrict__ d, int n,
                                 float* __restrict__ pmin, float* __restrict__ pmax) {
  __shared__ float smin[256], smax[256];
  float lmin = INFF, lmax = -INFF;
  for (int i = blockIdx.x * blockDim.x + threadIdx.x; i < n; i += gridDim.x * blockDim.x) {
    float v = d[i];
    lmin = fminf(lmin, v);
    lmax = fmaxf(lmax, v);
  }
  smin[threadIdx.x] = lmin; smax[threadIdx.x] = lmax;
  __syncthreads();
  for (int s = 128; s > 0; s >>= 1) {
    if ((int)threadIdx.x < s) {
      smin[threadIdx.x] = fminf(smin[threadIdx.x], smin[threadIdx.x + s]);
      smax[threadIdx.x] = fmaxf(smax[threadIdx.x], smax[threadIdx.x + s]);
    }
    __syncthreads();
  }
  if (threadIdx.x == 0) { pmin[blockIdx.x] = smin[0]; pmax[blockIdx.x] = smax[0]; }
}

__global__ void minmax_final_k(const float* __restrict__ pmin, const float* __restrict__ pmax,
                               float* __restrict__ minmax) {
  __shared__ float smin[256], smax[256];
  smin[threadIdx.x] = pmin[threadIdx.x];
  smax[threadIdx.x] = pmax[threadIdx.x];
  __syncthreads();
  for (int s = 128; s > 0; s >>= 1) {
    if ((int)threadIdx.x < s) {
      smin[threadIdx.x] = fminf(smin[threadIdx.x], smin[threadIdx.x + s]);
      smax[threadIdx.x] = fmaxf(smax[threadIdx.x], smax[threadIdx.x + s]);
    }
    __syncthreads();
  }
  if (threadIdx.x == 0) { minmax[0] = smin[0]; minmax[1] = smax[0]; }
}

__global__ void normalize_k(float* __restrict__ d, int n, const float* __restrict__ minmax) {
  float mn = minmax[0];
  float inv = 1.0f / (minmax[1] - minmax[0]);
  for (int i = blockIdx.x * blockDim.x + threadIdx.x; i < n; i += gridDim.x * blockDim.x)
    d[i] = (d[i] - mn) * inv;
}

__global__ void fill_zero_k(float* __restrict__ p, int n) {
  for (int i = blockIdx.x * blockDim.x + threadIdx.x; i < n; i += gridDim.x * blockDim.x)
    p[i] = 0.0f;
}

// ---------------- DTW forward DP: anti-diagonal wavefront -------------------
// One block per batch, thread j owns column j; three rolling diagonals in LDS.
__global__ void __launch_bounds__(512)
dtw_forward_k(const float* __restrict__ dist, signed char* __restrict__ choices) {
  int b = blockIdx.x;
  int j = threadIdx.x;
  __shared__ float buf[3][Tn];
  buf[0][j] = INFF; buf[1][j] = INFF; buf[2][j] = INFF;
  __syncthreads();
  const float* db = dist + (size_t)b * Tn * Tn;
  signed char* cb = choices + (size_t)b * Tn * Tn;
  for (int d = 0; d < 2 * Tn - 1; ++d) {
    float* cur = buf[d % 3];
    float* p1  = buf[(d + 2) % 3];   // diagonal d-1
    float* p2  = buf[(d + 1) % 3];   // diagonal d-2
    int i = d - j;
    if (i >= 0 && i < Tn) {
      // hide next step's dist read behind this diagonal's barrier
      if (i + 1 < Tn)
        __builtin_prefetch(db + ((size_t)(i + 1) * Tn + j), 0, 0);
      float dij = db[(size_t)i * Tn + j];
      float cost;
      signed char ch;
      if (d == 0) {                  // (0,0)
        cost = dij; ch = -1;
      } else if (j == 0) {           // first column: from up
        cost = p1[0] + dij; ch = 2;
      } else {
        float pd   = p2[j - 1];      // (i-1, j-1)
        float left = p1[j - 1];      // (i,   j-1)
        float up   = p1[j];          // (i-1, j)  (INF when i==0, never written)
        float m = fminf(fminf(pd, left), up);
        ch = (pd == m) ? 0 : ((left == m) ? 1 : 2);  // tie-break: diag,left,up
        cost = m + dij;
      }
      cur[j] = cost;
      cb[(size_t)i * Tn + j] = ch;
    }
    __syncthreads();
  }
}

// ---------------- backtrack: one lane per batch -----------------------------
__global__ void dtw_backtrack_k(const signed char* __restrict__ choices,
                                const int* __restrict__ real_length,
                                float* __restrict__ path) {
  int b = threadIdx.x;
  if (b >= Bn) return;
  const signed char* cb = choices + (size_t)b * Tn * Tn;
  float* pb = path + (size_t)b * Tn * Tn;
  int L = real_length[b];
  int ix = L - 1, iy = L - 1;
  pb[(size_t)ix * Tn + iy] = 1.0f;
  for (int s = 0; s < 2 * Tn - 2; ++s) {
    bool active = (ix > 0) || (iy > 0);
    signed char c = cb[(size_t)ix * Tn + iy];
    int nix = (c == 1) ? ix : ix - 1;
    int niy = (c == 2) ? iy : iy - 1;
    if (!active) { nix = ix; niy = iy; }
    if (active) pb[(size_t)nix * Tn + niy] = 1.0f;
    ix = nix; iy = niy;
  }
}

// ---------------------------------------------------------------------------
extern "C" void kernel_launch(void* const* d_in, const int* in_sizes, int n_in,
                              void* d_out, int out_size, void* d_ws, size_t ws_size,
                              hipStream_t stream) {
  (void)in_sizes; (void)n_in; (void)out_size; (void)ws_size;
  const float* vec   = (const float*)d_in[0];
  const float* music = (const float*)d_in[1];
  const int*   rlen  = (const int*)d_in[2];
  const float* qw1 = (const float*)d_in[3];  const float* qb1 = (const float*)d_in[4];
  const float* qw2 = (const float*)d_in[5];  const float* qb2 = (const float*)d_in[6];
  const float* qw3 = (const float*)d_in[7];  const float* qb3 = (const float*)d_in[8];
  const float* kw1 = (const float*)d_in[9];  const float* kb1 = (const float*)d_in[10];
  const float* kw2 = (const float*)d_in[11]; const float* kb2 = (const float*)d_in[12];
  const float* kw3 = (const float*)d_in[13]; const float* kb3 = (const float*)d_in[14];

  const size_t BHT = (size_t)Bn * Hn * Tn;   // 524288
  const size_t BTT = (size_t)Bn * Tn * Tn;   // 2097152

  float* path = (float*)d_out;               // first output
  float* dist = (float*)d_out + BTT;         // second output (normalized dist)

  char* w8 = (char*)d_ws;
  float*       buf0    = (float*)w8;        w8 += BHT * 4;
  float*       buf1    = (float*)w8;        w8 += BHT * 4;
  float*       qlat    = (float*)w8;        w8 += BHT * 4;
  float*       klat    = (float*)w8;        w8 += BHT * 4;
  _Float16*    qswz    = (_Float16*)w8;     w8 += BHT * 2;   // B-image fragments
  _Float16*    kswz    = (_Float16*)w8;     w8 += BHT * 2;   // A-image fragments
  float*       qnorm   = (float*)w8;        w8 += (size_t)Bn * Tn * 4;
  float*       knorm   = (float*)w8;        w8 += (size_t)Bn * Tn * 4;
  signed char* choices = (signed char*)w8;  w8 += BTT;
  float*       pmin    = (float*)w8;        w8 += 256 * 4;
  float*       pmax    = (float*)w8;        w8 += 256 * 4;
  float*       minmax  = (float*)w8;        w8 += 2 * 4;

  auto cdiv = [](int a, int b) { return (a + b - 1) / b; };

  // motion encoder: vec -> buf0 -> buf1 -> qlat
  conv_block_k<<<cdiv(Bn * CQn * Tn, 256), 256, 0, stream>>>(vec,  qw1, qb1, buf0, CQn, CQn);
  conv_block_k<<<cdiv(Bn * Hn  * Tn, 256), 256, 0, stream>>>(buf0, qw2, qb2, buf1, CQn, Hn);
  conv_block_k<<<cdiv(Bn * Hn  * Tn, 256), 256, 0, stream>>>(buf1, qw3, qb3, qlat, Hn,  Hn);
  // music encoder: music -> buf0 -> buf1 -> klat
  conv_block_k<<<cdiv(Bn * CKn * Tn, 256), 256, 0, stream>>>(music, kw1, kb1, buf0, CKn, CKn);
  conv_block_k<<<cdiv(Bn * Hn  * Tn, 256), 256, 0, stream>>>(buf0,  kw2, kb2, buf1, CKn, Hn);
  conv_block_k<<<cdiv(Bn * Hn  * Tn, 256), 256, 0, stream>>>(buf1,  kw3, kb3, klat, Hn,  Hn);

  norms_k<<<cdiv(Bn * Tn, 256), 256, 0, stream>>>(qlat, klat, qnorm, knorm);
  swizzle_frag_k<<<cdiv(Bn * 32 * 4 * 32, 256), 256, 0, stream>>>(klat, kswz, 0); // A image
  swizzle_frag_k<<<cdiv(Bn * 32 * 4 * 32, 256), 256, 0, stream>>>(qlat, qswz, 1); // B image

  // 8 batches * 32 M-tiles * 8 N-strips = 2048 waves, 4 waves per block
  dist_wmma_k<<<512, 128, 0, stream>>>(kswz, qswz, knorm, qnorm, dist);

  minmax_partial_k<<<256, 256, 0, stream>>>(dist, (int)BTT, pmin, pmax);
  minmax_final_k<<<1, 256, 0, stream>>>(pmin, pmax, minmax);
  normalize_k<<<2048, 256, 0, stream>>>(dist, (int)BTT, minmax);

  fill_zero_k<<<2048, 256, 0, stream>>>(path, (int)BTT);
  dtw_forward_k<<<Bn, 512, 0, stream>>>(dist, choices);
  dtw_backtrack_k<<<1, 32, 0, stream>>>(choices, rlen, path);
}
